// Head_9199819948631
// MI455X (gfx1250) — matrix operand order repaired and checked by reading, not compiled, not run on hardware
//
#include <hip/hip_runtime.h>
#include <hip/hip_bf16.h>
#include <stddef.h>
#include <stdint.h>

// ---------------------------------------------------------------------------
// Fused causal attention head for MI455X (gfx1250, wave32, WMMA bf16)
//   proj:  qkv = x @ [Wq;Wk;Wv]^T      (bf16 WMMA, fp32 acc)
//   attn:  flash-attention, online softmax, async K-tile staging to LDS
// ---------------------------------------------------------------------------

#define BATCH 256
#define TSEQ  256
#define CEMB  384
#define HS    64

typedef __attribute__((ext_vector_type(16))) __bf16        v16bf;
typedef __attribute__((ext_vector_type(2)))  __bf16        v2bf;
typedef __attribute__((ext_vector_type(8)))  float         v8f;
typedef __attribute__((ext_vector_type(8)))  unsigned int  v8u;

// Native f32 -> bf16 (RNE); backend pairs these into v_cvt_pk_bf16_f32.
__device__ __forceinline__ unsigned short f2bf(float f) {
  __bf16 b = (__bf16)f;
  return __builtin_bit_cast(unsigned short, b);
}
__device__ __forceinline__ unsigned int cvt_pk_bf16(float lo, float hi) {
  v2bf p;
  p[0] = (__bf16)lo;
  p[1] = (__bf16)hi;
  return __builtin_bit_cast(unsigned int, p);
}

__device__ __forceinline__ v8u mk8(uint4 a, uint4 b) {
  v8u r;
  r[0] = a.x; r[1] = a.y; r[2] = a.z; r[3] = a.w;
  r[4] = b.x; r[5] = b.y; r[6] = b.z; r[7] = b.w;
  return r;
}

// A-fragment (16x32 bf16), ISA 7.12.2: lane row = lane&15; per lane-half the
// K pairs are 8 contiguous elements at lh*8 (v0..3) and 16+lh*8 (v4..7):
// exactly two b128 loads from the lane's row pointer.
__device__ __forceinline__ v8u load_afrag(const unsigned short* rowp, int kc, int lh) {
  const uint4* p0 = (const uint4*)(rowp + kc * 32 + lh * 8);
  const uint4* p1 = (const uint4*)(rowp + kc * 32 + 16 + lh * 8);
  return mk8(*p0, *p1);
}
// B-fragment (32x16 bf16): lane col = lane&15; per lane-half the K pairs are
// 8 contiguous elements at lh*8 (v0..3) and 16+lh*8 (v4..7): two b128 loads.
__device__ __forceinline__ v8u load_bfrag(const unsigned short* colp, int kc, int lh) {
  const uint4* p0 = (const uint4*)(colp + kc * 32 + lh * 8);
  const uint4* p1 = (const uint4*)(colp + kc * 32 + 16 + lh * 8);
  return mk8(*p0, *p1);
}

__device__ __forceinline__ v8f wmma_bf16(v8u a, v8u b, v8f c) {
  return __builtin_amdgcn_wmma_f32_16x16x32_bf16(
      /*neg_a=*/false, __builtin_bit_cast(v16bf, a),
      /*neg_b=*/false, __builtin_bit_cast(v16bf, b),
      /*c_mod=*/(short)0, c, /*reuse_a=*/false, /*reuse_b=*/false);
}

// ---------------------------------------------------------------------------
// Kernel 0: fp32 weights -> concatenated bf16 weight  wcat[j][c], j: q|k|v
// ---------------------------------------------------------------------------
__global__ __launch_bounds__(256) void wconv_kernel(const float* __restrict__ Wq,
                                                    const float* __restrict__ Wk,
                                                    const float* __restrict__ Wv,
                                                    unsigned short* __restrict__ wcat) {
  int idx = blockIdx.x * 256 + threadIdx.x;
  if (idx >= 192 * CEMB) return;
  int j = idx / CEMB, c = idx % CEMB;
  const float* src = (j < 64) ? Wq : (j < 128 ? Wk : Wv);
  wcat[idx] = f2bf(src[(j & 63) * CEMB + c]);
}

// ---------------------------------------------------------------------------
// Kernel 1: projection GEMM  [65536 x 384] * [384 x 192] -> q,k,v (bf16)
// 256 threads = 8 waves; wave owns 16 rows x 192 cols (12 C/D tiles).
// B fragments are software-pipelined one tile ahead of the WMMA consuming
// the current one, so the per-tile s_wait overlaps matrix math.
// ---------------------------------------------------------------------------
__global__ __launch_bounds__(256) void proj_kernel(const float* __restrict__ x,
                                                   const unsigned short* __restrict__ wcat,
                                                   unsigned short* __restrict__ qb,
                                                   unsigned short* __restrict__ kb,
                                                   unsigned short* __restrict__ vb) {
  const int tid = threadIdx.x;
  const int w   = tid >> 5;
  const int l   = tid & 31;
  const int ln  = l & 15;
  const int lh  = l >> 4;
  const int row0 = blockIdx.x * 128 + w * 16;       // 16 rows per wave
  const float* xr = x + (size_t)(row0 + ln) * CEMB; // A row for this lane

  v8f acc[12];
  #pragma unroll
  for (int nt = 0; nt < 12; ++nt) acc[nt] = (v8f){0,0,0,0,0,0,0,0};

  #pragma unroll
  for (int kc = 0; kc < 12; ++kc) {
    // A fragment from fp32 x: four float4 loads + packed bf16 converts.
    float4 g0 = *(const float4*)(xr + kc * 32 + lh * 8);
    float4 g1 = *(const float4*)(xr + kc * 32 + lh * 8 + 4);
    float4 g2 = *(const float4*)(xr + kc * 32 + 16 + lh * 8);
    float4 g3 = *(const float4*)(xr + kc * 32 + 16 + lh * 8 + 4);
    v8u aX;
    aX[0] = cvt_pk_bf16(g0.x, g0.y);
    aX[1] = cvt_pk_bf16(g0.z, g0.w);
    aX[2] = cvt_pk_bf16(g1.x, g1.y);
    aX[3] = cvt_pk_bf16(g1.z, g1.w);
    aX[4] = cvt_pk_bf16(g2.x, g2.y);
    aX[5] = cvt_pk_bf16(g2.z, g2.w);
    aX[6] = cvt_pk_bf16(g3.x, g3.y);
    aX[7] = cvt_pk_bf16(g3.z, g3.w);

    // One-deep pipeline over the 12 N-tiles.
    v8u bW = load_bfrag(wcat + (size_t)ln * CEMB, kc, lh);       // nt = 0
    #pragma unroll
    for (int nt = 0; nt < 12; ++nt) {
      v8u bWnext = bW;
      if (nt < 11) {
        int jn = (nt + 1) * 16 + ln;
        bWnext = load_bfrag(wcat + (size_t)jn * CEMB, kc, lh);
      }
      acc[nt] = wmma_bf16(aX, bW, acc[nt]);
      bW = bWnext;
    }
  }

  // C/D layout: VGPR r -> row r (lanes 0-15) / r+8 (lanes 16-31); col = lane&15
  #pragma unroll
  for (int nt = 0; nt < 12; ++nt) {
    int j = nt * 16 + ln;
    unsigned short* dst = (j < 64) ? qb : (j < 128 ? kb : vb);
    int jj = j & 63;
    #pragma unroll
    for (int r = 0; r < 8; ++r) {
      int grow = row0 + r + 8 * lh;
      dst[(size_t)grow * HS + jj] = f2bf(acc[nt][r]);
    }
  }
}

// ---------------------------------------------------------------------------
// Kernel 2: flash attention. Block = 128 thr = 4 waves = one (b, 64-row qtile)
// K tile staged with global_load_async_to_lds_b128 (ASYNCcnt), V transposed.
// ---------------------------------------------------------------------------
__global__ __launch_bounds__(128) void attn_kernel(const unsigned short* __restrict__ qb,
                                                   const unsigned short* __restrict__ kb,
                                                   const unsigned short* __restrict__ vb,
                                                   float* __restrict__ out) {
  __shared__ __align__(16) unsigned short Ks[64 * HS];    // K tile [kpos][feat]
  __shared__ __align__(16) unsigned short Vt[HS * 64];    // V tile [feat][kpos]
  __shared__ __align__(16) unsigned short Pb[4][16 * 64]; // per-wave P staging

  const int b  = blockIdx.x >> 2;
  const int qt = blockIdx.x & 3;
  const int q0 = qt * 64;
  const int tid = threadIdx.x;
  const int w  = tid >> 5;
  const int l  = tid & 31;
  const int ln = l & 15;
  const int lh = l >> 4;

  // Q A-fragments (2 K-chunks over head dim 64), loaded once from global
  const int qrow = q0 + w * 16 + ln;
  const unsigned short* qrp = qb + (size_t)(b * TSEQ + qrow) * HS;
  v8u aQ[2];
  #pragma unroll
  for (int kc = 0; kc < 2; ++kc) aQ[kc] = load_afrag(qrp, kc, lh);

  v8f O[4];
  #pragma unroll
  for (int nt = 0; nt < 4; ++nt) O[nt] = (v8f){0,0,0,0,0,0,0,0};
  float m_run[8], l_run[8];
  #pragma unroll
  for (int r = 0; r < 8; ++r) { m_run[r] = -1e30f; l_run[r] = 0.0f; }

  const unsigned ldsK = (unsigned)(uintptr_t)(&Ks[0]);   // low 32 bits = LDS offset

  const int ktE = qt + 1;                      // causal: tiles at/before q-tile
  for (int kt = 0; kt < ktE; ++kt) {
    __syncthreads();                           // prev iteration's LDS reads done

    // ---- K tile: async global->LDS, 128 lanes x 16B x 4 passes = 8 KB ----
    {
      const unsigned short* kg = kb + (size_t)(b * TSEQ + kt * 64) * HS;
      #pragma unroll
      for (int p = 0; p < 4; ++p) {
        int e = (p * 128 + tid) * 8;           // ushort index
        unsigned la = ldsK + (unsigned)(e * 2);
        const void* ga = (const void*)(kg + e);
        asm volatile("global_load_async_to_lds_b128 %0, %1, off"
                     :: "v"(la), "v"(ga)
                     : "memory");
      }
    }

    // ---- prefetch next K/V tiles into cache while this tile computes ----
    if (kt + 1 < ktE) {
      const char* nk = (const char*)(kb + (size_t)(b * TSEQ + (kt + 1) * 64) * HS);
      const char* nv = (const char*)(vb + (size_t)(b * TSEQ + (kt + 1) * 64) * HS);
      __builtin_prefetch(nk + tid * 64, 0, 1);
      __builtin_prefetch(nv + tid * 64, 0, 1);
    }

    // ---- V tile: load row-major, store transposed into LDS ----
    {
      const unsigned int* vg = (const unsigned int*)(vb + (size_t)(b * TSEQ + kt * 64) * HS);
      for (int i = tid; i < 64 * (HS / 2); i += 128) {
        unsigned int d = vg[i];
        int row = i >> 5;                      // key position (local)
        int fp  = (i & 31) * 2;                // feature pair
        Vt[fp * 64 + row]       = (unsigned short)(d & 0xFFFFu);
        Vt[(fp + 1) * 64 + row] = (unsigned short)(d >> 16);
      }
    }

    asm volatile("s_wait_asynccnt 0x0" ::: "memory");
    __syncthreads();

    // ---- S = Q K^T (4 N-tiles x 2 K-chunks), K frags pipelined one ahead ----
    v8f S[4];
    {
      v8u bK = load_bfrag(Ks + (0 * 16 + ln) * HS, 0, lh);
      #pragma unroll
      for (int nt = 0; nt < 4; ++nt) {
        v8f acc = (v8f){0,0,0,0,0,0,0,0};
        #pragma unroll
        for (int kc = 0; kc < 2; ++kc) {
          v8u bKnext = bK;
          if (!(nt == 3 && kc == 1)) {
            int nn = (kc == 1) ? nt + 1 : nt;
            int cc = (kc == 1) ? 0 : 1;
            bKnext = load_bfrag(Ks + (nn * 16 + ln) * HS, cc, lh);
          }
          acc = wmma_bf16(aQ[kc], bK, acc);
          bK = bKnext;
        }
        S[nt] = acc;
      }
    }

    // ---- scale + causal mask + online softmax (rows match C layout) ----
    #pragma unroll
    for (int r = 0; r < 8; ++r) {
      int grow = q0 + w * 16 + r + 8 * lh;
      float mx = -1e30f;
      #pragma unroll
      for (int nt = 0; nt < 4; ++nt) {
        int gcol = kt * 64 + nt * 16 + ln;
        float s = S[nt][r] * 0.125f;           // 1/sqrt(64)
        s = (gcol <= grow) ? s : -1e30f;
        S[nt][r] = s;
        mx = fmaxf(mx, s);
      }
      #pragma unroll
      for (int m = 1; m < 16; m <<= 1) mx = fmaxf(mx, __shfl_xor(mx, m, 32));
      float mnew  = fmaxf(m_run[r], mx);
      float alpha = __expf(m_run[r] - mnew);
      m_run[r] = mnew;
      float rs = 0.0f;
      #pragma unroll
      for (int nt = 0; nt < 4; ++nt) {
        float p = __expf(S[nt][r] - mnew);
        S[nt][r] = p;
        rs += p;
      }
      #pragma unroll
      for (int m = 1; m < 16; m <<= 1) rs += __shfl_xor(rs, m, 32);
      l_run[r] = l_run[r] * alpha + rs;
      #pragma unroll
      for (int nt = 0; nt < 4; ++nt) O[nt][r] *= alpha;
    }

    // ---- P (C layout) -> LDS -> A layout, then O += P V ----
    unsigned short* pw = Pb[w];
    #pragma unroll
    for (int nt = 0; nt < 4; ++nt)
      #pragma unroll
      for (int r = 0; r < 8; ++r)
        pw[(r + 8 * lh) * 64 + nt * 16 + ln] = f2bf(S[nt][r]);

    v8u aP[2];
    #pragma unroll
    for (int kc = 0; kc < 2; ++kc) aP[kc] = load_afrag(pw + ln * 64, kc, lh);

    #pragma unroll
    for (int nt = 0; nt < 4; ++nt) {
      int hc = nt * 16 + ln;                   // B col (lane) = head feature
      #pragma unroll
      for (int kc = 0; kc < 2; ++kc) {
        v8u bV = load_bfrag(Vt + hc * 64, kc, lh);
        O[nt] = wmma_bf16(aP[kc], bV, O[nt]);
      }
    }
  }

  // ---- epilogue: O / l, fp32 store ----
  #pragma unroll
  for (int r = 0; r < 8; ++r) {
    float inv  = 1.0f / l_run[r];
    int   grow = q0 + w * 16 + r + 8 * lh;
    #pragma unroll
    for (int nt = 0; nt < 4; ++nt)
      out[(size_t)(b * TSEQ + grow) * HS + nt * 16 + ln] = O[nt][r] * inv;
  }
}

// ---------------------------------------------------------------------------
// Host side
// ---------------------------------------------------------------------------
extern "C" void kernel_launch(void* const* d_in, const int* in_sizes, int n_in,
                              void* d_out, int out_size, void* d_ws, size_t ws_size,
                              hipStream_t stream) {
  // setup_inputs order: x, Wk, Wq, Wv   (note: Wk before Wq)
  const float* x  = (const float*)d_in[0];
  const float* Wk = (const float*)d_in[1];
  const float* Wq = (const float*)d_in[2];
  const float* Wv = (const float*)d_in[3];
  float* out = (float*)d_out;

  char* ws = (char*)d_ws;
  const size_t qkv_bytes = (size_t)BATCH * TSEQ * HS * sizeof(unsigned short); // 8 MB each
  unsigned short* qb   = (unsigned short*)(ws);
  unsigned short* kb   = (unsigned short*)(ws + qkv_bytes);
  unsigned short* vb   = (unsigned short*)(ws + 2 * qkv_bytes);
  unsigned short* wcat = (unsigned short*)(ws + 3 * qkv_bytes);               // 144 KB

  // 0) weights -> bf16
  wconv_kernel<<<(192 * CEMB + 255) / 256, 256, 0, stream>>>(Wq, Wk, Wv, wcat);
  // 1) q,k,v projections (WMMA GEMM)
  proj_kernel<<<(BATCH * TSEQ) / 128, 256, 0, stream>>>(x, wcat, qb, kb, vb);
  // 2) fused flash attention (WMMA + async LDS staging)
  attn_kernel<<<BATCH * (TSEQ / 64), 128, 0, stream>>>(qb, kb, vb, out);
}
